// myMultiheadAttention_90323162235350
// MI455X (gfx1250) — compile-verified, matrix operand order checked by reading, not run on hardware
//
#include <hip/hip_runtime.h>
#include <hip/hip_bf16.h>

// ---------------------------------------------------------------------------
// Multi-head attention, fp32 in/out, bf16 WMMA compute (gfx1250 / CDNA5)
// B=2, S=2048, E=1024, H=16, D=64
// Pipeline:
//   k_cvt      : fp32 -> bf16 one-time conversion (bandwidth bound)
//   k_proj_qk  : C^T = Wq/k * X^T  -> [B,H,S,D] bf16 (Q scaled 1/sqrt(D))
//   k_proj_v   : C   = X * Wv^T    -> V^T [B,H,D,S] bf16
//   k_attn     : flash attention, scores transposed (S^T = K*Q^T),
//                K/V blocks staged in LDS by the Tensor Data Mover (TDM),
//                double-buffered, synced with s_wait_tensorcnt.
//   k_outproj  : Y^T = Wout * O^T  -> fp32 output
// ---------------------------------------------------------------------------

#define BB 2
#define SS 2048
#define EE 1024
#define HH 16
#define DD 64
#define BS (BB * SS)   // 4096 tokens

typedef __attribute__((ext_vector_type(16))) __bf16 v16bf;
typedef __attribute__((ext_vector_type(8)))  __bf16 v8bf;
typedef __attribute__((ext_vector_type(8)))  float  v8f;
typedef unsigned int u32x4 __attribute__((ext_vector_type(4)));
typedef int          i32x4 __attribute__((ext_vector_type(4)));
typedef int          i32x8 __attribute__((ext_vector_type(8)));

#if __has_builtin(__builtin_amdgcn_tensor_load_to_lds) && \
    __has_builtin(__builtin_amdgcn_s_wait_tensorcnt)
#define HAVE_TDM 1
#else
#define HAVE_TDM 0
#endif

__device__ __forceinline__ __bf16 f2bf(float f) { return (__bf16)f; }

__device__ __forceinline__ v8f wmma_bf16(v16bf a, v16bf b, v8f c) {
  // D = A(16x32 bf16) * B(32x16 bf16) + C(16x16 f32)
  return __builtin_amdgcn_wmma_f32_16x16x32_bf16(
      /*neg_a=*/false, a, /*neg_b=*/false, b,
      /*c_mod=*/(short)0, c, /*reuse_a=*/false, /*reuse_b=*/false);
}

// One 16-bit A/B-layout fragment from a bf16 row contiguous in K.
// ISA layout: lane half selects K runs [8h..8h+7] and [16+8h..16+8h+7].
__device__ __forceinline__ v16bf load_frag_bf16(const __bf16* __restrict__ src,
                                                int half) {
  union { v16bf v; uint4 q[2]; } u;
  u.q[0] = *(const uint4*)(src + 8 * half);
  u.q[1] = *(const uint4*)(src + 16 + 8 * half);
  return u.v;
}

#if HAVE_TDM
// Issue a TDM 2D tile load (bf16 elements) into LDS.  D# built per ISA ch.8:
//  group0: count=1 | lds_addr | global_addr[56:0] | type=2
//  group1: data_size=1(2B) | tensor_dim0/1 | tile_dim0/1 | tensor_dim0_stride
// Dims in elements; rows packed contiguously in LDS.
// Toolchain note: this clang exposes the 6-arg builtin
//   (u32x4 g0, i32x8 g1, i32x4 g2, i32x4 g3, i32x8 g4, i32 cpol)
__device__ __forceinline__ void tdm_load_2d(unsigned lds_off, const void* gptr,
                                            unsigned tile_d0, unsigned tile_d1,
                                            unsigned tensor_d0,
                                            unsigned tensor_d1,
                                            unsigned stride_d0) {
  const unsigned long long ga = (unsigned long long)(uintptr_t)gptr;
  u32x4 g0;
  g0.x = 1u;                                      // count=1, user descriptor
  g0.y = lds_off;                                 // lds_addr (bytes)
  g0.z = (unsigned)(ga & 0xFFFFFFFFu);            // global_addr[31:0]
  g0.w = (unsigned)((ga >> 32) & 0x01FFFFFFu) |   // global_addr[56:32]
         (2u << 30);                              // type = 2 ("image")
  i32x8 g1;
  g1[0] = (int)(1u << 16);                        // wg_mask=0, data_size=2B
  g1[1] = (int)((tensor_d0 & 0xFFFFu) << 16);     // tensor_dim0[15:0]
  g1[2] = (int)(((tensor_d0 >> 16) & 0xFFFFu) |
                ((tensor_d1 & 0xFFFFu) << 16));   // td0 hi | td1 lo
  g1[3] = (int)(((tensor_d1 >> 16) & 0xFFFFu) |
                (tile_d0 << 16));                 // td1 hi | tile_dim0
  g1[4] = (int)(tile_d1 & 0xFFFFu);               // tile_dim1 (tile_dim2=0)
  g1[5] = (int)stride_d0;                         // tensor_dim0_stride[31:0]
  g1[6] = 0;                                      // stride hi, dim1_stride=0
  g1[7] = 0;
  const i32x4 z4 = {0, 0, 0, 0};                  // groups 2/3 unused (2D)
  const i32x8 z8 = {0, 0, 0, 0, 0, 0, 0, 0};
  __builtin_amdgcn_tensor_load_to_lds(g0, g1, z4, z4, z8, 0);
}
#endif

// ---------------------------------------------------------------------------
// Kernel 0: fp32 -> bf16 bulk conversion (8 elements / thread)
// ---------------------------------------------------------------------------
__global__ __launch_bounds__(256) void k_cvt(const float* __restrict__ src,
                                             __bf16* __restrict__ dst, int n8) {
  const int idx = blockIdx.x * 256 + threadIdx.x;
  if (idx >= n8) return;
  const float4* s = (const float4*)src + (size_t)idx * 2;
  const float4 lo = s[0], hi = s[1];
  union { v8bf v; uint4 q; } u;
  u.v[0] = f2bf(lo.x); u.v[1] = f2bf(lo.y);
  u.v[2] = f2bf(lo.z); u.v[3] = f2bf(lo.w);
  u.v[4] = f2bf(hi.x); u.v[5] = f2bf(hi.y);
  u.v[6] = f2bf(hi.z); u.v[7] = f2bf(hi.w);
  *((uint4*)dst + idx) = u.q;
}

// ---------------------------------------------------------------------------
// Kernel 1: Q/K in-projection, computed transposed:  C^T = W * X^T
// Wave: 2x4 WMMA tiles.  WG: 64 feat x 256 tok.  grid: (BS/256, EE/64, 2)
// ---------------------------------------------------------------------------
__global__ __launch_bounds__(256) void k_proj_qk(
    const __bf16* __restrict__ xq, const __bf16* __restrict__ xk,
    const __bf16* __restrict__ wb, const float* __restrict__ in_b,
    __bf16* __restrict__ qws, __bf16* __restrict__ kws) {
  const int which = blockIdx.z;                        // 0 = Q, 1 = K
  const __bf16* X = which ? xk : xq;
  const __bf16* W = wb + (size_t)which * EE * EE;      // packed rows: q,k,v
  const float*  bv = in_b + which * EE;
  __bf16* out = which ? kws : qws;
  const float scl = which ? 1.0f : 0.125f;             // 1/sqrt(64) on Q

  const int lane = threadIdx.x & 31;
  const int wid  = threadIdx.x >> 5;
  const int half = lane >> 4;
  const int ln   = lane & 15;
  const int feat0 = blockIdx.y * 64 + (wid & 1) * 32;
  const int tok0  = blockIdx.x * 256 + (wid >> 1) * 64;

  v8f acc[2][4] = {};
  for (int kk = 0; kk < EE; kk += 32) {
    v16bf a[2], b[4];
#pragma unroll
    for (int i = 0; i < 2; ++i)
      a[i] = load_frag_bf16(W + (size_t)(feat0 + 16 * i + ln) * EE + kk, half);
#pragma unroll
    for (int j = 0; j < 4; ++j)
      b[j] = load_frag_bf16(X + (size_t)(tok0 + 16 * j + ln) * EE + kk, half);
#pragma unroll
    for (int i = 0; i < 2; ++i)
#pragma unroll
      for (int j = 0; j < 4; ++j)
        acc[i][j] = wmma_bf16(a[i], b[j], acc[i][j]);
  }

  // C^T layout: lane column = token, rows (VGPR r) = 8 consecutive features.
#pragma unroll
  for (int i = 0; i < 2; ++i) {
    const int f0 = feat0 + 16 * i + 8 * half;
    float bias[8];
#pragma unroll
    for (int r = 0; r < 8; ++r) bias[r] = bv[f0 + r];
    const int hh = f0 >> 6, d0 = f0 & 63;
#pragma unroll
    for (int j = 0; j < 4; ++j) {
      const int t  = tok0 + 16 * j + ln;
      const int bb = t >> 11, s = t & (SS - 1);
      union { v8bf v; uint4 q; } u;
#pragma unroll
      for (int r = 0; r < 8; ++r)
        u.v[r] = f2bf((acc[i][j][r] + bias[r]) * scl);
      *(uint4*)(out + ((((size_t)bb * HH + hh) * SS + s) * DD + d0)) = u.q;
    }
  }
}

// ---------------------------------------------------------------------------
// Kernel 2: V in-projection, normal C = X * W^T  -> V^T [B,H,D,S] bf16.
// grid: (EE/256, BS/64)
// ---------------------------------------------------------------------------
__global__ __launch_bounds__(256) void k_proj_v(
    const __bf16* __restrict__ xv, const __bf16* __restrict__ wb,
    const float* __restrict__ in_b, __bf16* __restrict__ vt) {
  const __bf16* W  = wb + (size_t)2 * EE * EE;
  const float*  bv = in_b + 2 * EE;

  const int lane = threadIdx.x & 31;
  const int wid  = threadIdx.x >> 5;
  const int half = lane >> 4;
  const int ln   = lane & 15;
  const int tok0  = blockIdx.y * 64 + (wid & 1) * 32;    // M rows
  const int feat0 = blockIdx.x * 256 + (wid >> 1) * 64;  // N cols

  v8f acc[2][4] = {};
  for (int kk = 0; kk < EE; kk += 32) {
    v16bf a[2], b[4];
#pragma unroll
    for (int i = 0; i < 2; ++i)
      a[i] = load_frag_bf16(xv + (size_t)(tok0 + 16 * i + ln) * EE + kk, half);
#pragma unroll
    for (int j = 0; j < 4; ++j)
      b[j] = load_frag_bf16(W + (size_t)(feat0 + 16 * j + ln) * EE + kk, half);
#pragma unroll
    for (int i = 0; i < 2; ++i)
#pragma unroll
      for (int j = 0; j < 4; ++j)
        acc[i][j] = wmma_bf16(a[i], b[j], acc[i][j]);
  }

  // C layout: lane column = feature, rows = 8 consecutive tokens.
#pragma unroll
  for (int j = 0; j < 4; ++j) {
    const int f  = feat0 + 16 * j + ln;
    const int hh = f >> 6, d = f & 63;
    const float bias = bv[f];
#pragma unroll
    for (int i = 0; i < 2; ++i) {
      const int t0 = tok0 + 16 * i + 8 * half;
      const int bb = t0 >> 11, s0 = t0 & (SS - 1);
      union { v8bf v; uint4 q; } u;
#pragma unroll
      for (int r = 0; r < 8; ++r) u.v[r] = f2bf(acc[i][j][r] + bias);
      *(uint4*)(vt + ((((size_t)bb * HH + hh) * DD + d) * SS + s0)) = u.q;
    }
  }
}

// ---------------------------------------------------------------------------
// Kernel 3: flash attention per (b,h).  Scores computed TRANSPOSED
// (S^T = K_blk * Q^T): the score C-layout == A-layout of P for P@V.
// Wave: 32 queries (2 tiles).  WG: 8 waves = 256 queries.
// K/V 32-key blocks staged into double-buffered LDS by the TDM (wave 0
// issues tensor_load_to_lds; s_wait_tensorcnt + one barrier per block).
// grid: (SS/256, HH, BB)
// ---------------------------------------------------------------------------
__global__ __launch_bounds__(256) void k_attn(
    const __bf16* __restrict__ qws, const __bf16* __restrict__ kws,
    const __bf16* __restrict__ vt, __bf16* __restrict__ ows) {
  __shared__ __bf16 Ks[2][32 * DD];   // [buf][key][d]
  __shared__ __bf16 Vs[2][DD * 32];   // [buf][d][key]

  const int hh = blockIdx.y, bb = blockIdx.z;
  const int lane = threadIdx.x & 31;
  const int wid  = threadIdx.x >> 5;
  const int half = lane >> 4;
  const int ln   = lane & 15;

  const size_t head = ((size_t)bb * HH + hh) * SS * DD;
  const __bf16* Qh = qws + head;   // [s][d]
  const __bf16* Kh = kws + head;   // [s][d]
  const __bf16* Vh = vt + head;    // [d][s]
  const int q0 = blockIdx.x * 256 + wid * 32;

  // Q as the fixed B operand of S^T: lane column = query, K-dim = d.
  v16bf qb[2][2];
#pragma unroll
  for (int qt = 0; qt < 2; ++qt)
#pragma unroll
    for (int c = 0; c < 2; ++c)
      qb[qt][c] =
          load_frag_bf16(Qh + (size_t)(q0 + 16 * qt + ln) * DD + 32 * c, half);

  float m_i[2] = {-1e30f, -1e30f}, l_i[2] = {0.0f, 0.0f};
  v8f o[2][4] = {};                // per query tile: 16q x 64d, 4 d-tiles

#if HAVE_TDM
  if (wid == 0) {                  // prologue: fill buffer 0
    tdm_load_2d((unsigned)(uintptr_t)&Ks[0][0], Kh, DD, 32, DD, SS, DD);
    tdm_load_2d((unsigned)(uintptr_t)&Vs[0][0], Vh, 32, DD, SS, DD, SS);
  }
#endif

  for (int kb = 0, it = 0; kb < SS; kb += 32, ++it) {
    const int cur = it & 1;
#if HAVE_TDM
    if (wid == 0) __builtin_amdgcn_s_wait_tensorcnt(0);
    __syncthreads();               // buf[cur] ready; buf[cur^1] drained
    if (wid == 0 && kb + 32 < SS) {
      tdm_load_2d((unsigned)(uintptr_t)&Ks[cur ^ 1][0],
                  Kh + (size_t)(kb + 32) * DD, DD, 32, DD, SS - kb - 32, DD);
      tdm_load_2d((unsigned)(uintptr_t)&Vs[cur ^ 1][0],
                  Vh + kb + 32, 32, DD, SS - kb - 32, DD, SS);
    }
#else
    __syncthreads();
    {
      const int t = threadIdx.x;   // manual staging fallback
      *(uint4*)(&Ks[cur][0] + t * 8) =
          *(const uint4*)(Kh + (size_t)(kb + (t >> 3)) * DD + (t & 7) * 8);
      *(uint4*)(&Vs[cur][0] + t * 8) =
          *(const uint4*)(Vh + (size_t)(t >> 2) * SS + kb + (t & 3) * 8);
    }
    __syncthreads();
#endif
    const __bf16* ksb = &Ks[cur][0];
    const __bf16* vsb = &Vs[cur][0];

    // K / V fragments once per block, shared by both query tiles.
    v16bf ka[2][2], vb[4];
#pragma unroll
    for (int t2 = 0; t2 < 2; ++t2)
#pragma unroll
      for (int c = 0; c < 2; ++c)
        ka[t2][c] = load_frag_bf16(ksb + (16 * t2 + ln) * DD + 32 * c, half);
#pragma unroll
    for (int t2 = 0; t2 < 4; ++t2)
      vb[t2] = load_frag_bf16(vsb + (ln + 16 * t2) * 32, half);

#pragma unroll
    for (int qt = 0; qt < 2; ++qt) {
      // S^T tiles: st[0] = keys kb..kb+15, st[1] = keys kb+16..kb+31
      v8f st[2] = {};
#pragma unroll
      for (int t2 = 0; t2 < 2; ++t2)
#pragma unroll
        for (int c = 0; c < 2; ++c)
          st[t2] = wmma_bf16(ka[t2][c], qb[qt][c], st[t2]);

      // Online softmax (per-lane 16 values + one shfl_xor(16)).
      float mb = -1e30f;
#pragma unroll
      for (int r = 0; r < 8; ++r)
        mb = fmaxf(mb, fmaxf(st[0][r], st[1][r]));
      mb = fmaxf(mb, __shfl_xor(mb, 16, 32));
      const float m_new = fmaxf(m_i[qt], mb);

      float p0[8], p1[8], ls = 0.0f;
#pragma unroll
      for (int r = 0; r < 8; ++r) {
        p0[r] = __expf(st[0][r] - m_new);
        p1[r] = __expf(st[1][r] - m_new);
        ls += p0[r] + p1[r];
      }
      ls += __shfl_xor(ls, 16, 32);
      const float alpha = __expf(m_i[qt] - m_new);
      l_i[qt] = l_i[qt] * alpha + ls;
      m_i[qt] = m_new;

      // P fragment: C-layout of (st0,st1) == A-layout of P.
      v16bf pa;
#pragma unroll
      for (int e = 0; e < 8; ++e)  pa[e] = f2bf(p0[e]);
#pragma unroll
      for (int e = 8; e < 16; ++e) pa[e] = f2bf(p1[e - 8]);

      // Rescale running O by alpha of the query each accumulator row holds.
#pragma unroll
      for (int r = 0; r < 8; ++r) {
        const float av = __shfl(alpha, r + 8 * half, 32);
#pragma unroll
        for (int t2 = 0; t2 < 4; ++t2) o[qt][t2][r] *= av;
      }

      // O += P @ V  (K-dim = 32 keys: one WMMA per d-tile)
#pragma unroll
      for (int t2 = 0; t2 < 4; ++t2)
        o[qt][t2] = wmma_bf16(pa, vb[t2], o[qt][t2]);
    }
  }

  // Normalize and store bf16 O into token-major [BS][EE] workspace.
#pragma unroll
  for (int qt = 0; qt < 2; ++qt)
#pragma unroll
    for (int r = 0; r < 8; ++r) {
      const float lq  = __shfl(l_i[qt], r + 8 * half, 32);
      const float inv = 1.0f / lq;
      const size_t row =
          ((size_t)bb * SS + q0 + 16 * qt + 8 * half + r) * EE + hh * DD;
#pragma unroll
      for (int t2 = 0; t2 < 4; ++t2)
        ows[row + 16 * t2 + ln] = f2bf(o[qt][t2][r] * inv);
    }
}

// ---------------------------------------------------------------------------
// Kernel 4: out-projection  Y^T = Wout * O^T ; fp32 output, contiguous stores.
// grid: (BS/256, EE/64)
// ---------------------------------------------------------------------------
__global__ __launch_bounds__(256) void k_outproj(
    const __bf16* __restrict__ ows, const __bf16* __restrict__ wob,
    const float* __restrict__ out_b, float* __restrict__ out) {
  const int lane = threadIdx.x & 31;
  const int wid  = threadIdx.x >> 5;
  const int half = lane >> 4;
  const int ln   = lane & 15;
  const int feat0 = blockIdx.y * 64 + (wid & 1) * 32;
  const int tok0  = blockIdx.x * 256 + (wid >> 1) * 64;

  v8f acc[2][4] = {};
  for (int kk = 0; kk < EE; kk += 32) {
    v16bf a[2], b[4];
#pragma unroll
    for (int i = 0; i < 2; ++i)
      a[i] = load_frag_bf16(wob + (size_t)(feat0 + 16 * i + ln) * EE + kk, half);
#pragma unroll
    for (int j = 0; j < 4; ++j)
      b[j] = load_frag_bf16(ows + (size_t)(tok0 + 16 * j + ln) * EE + kk, half);
#pragma unroll
    for (int i = 0; i < 2; ++i)
#pragma unroll
      for (int j = 0; j < 4; ++j)
        acc[i][j] = wmma_bf16(a[i], b[j], acc[i][j]);
  }

#pragma unroll
  for (int i = 0; i < 2; ++i) {
    const int f0 = feat0 + 16 * i + 8 * half;
    float bias[8];
#pragma unroll
    for (int r = 0; r < 8; ++r) bias[r] = out_b[f0 + r];
#pragma unroll
    for (int j = 0; j < 4; ++j) {
      const int t = tok0 + 16 * j + ln;
      float res[8];
#pragma unroll
      for (int r = 0; r < 8; ++r) res[r] = acc[i][j][r] + bias[r];
      float* dst = out + (size_t)t * EE + f0;
      *(float4*)(dst)     = *(float4*)(res);
      *(float4*)(dst + 4) = *(float4*)(res + 4);
    }
  }
}

// ---------------------------------------------------------------------------
extern "C" void kernel_launch(void* const* d_in, const int* in_sizes, int n_in,
                              void* d_out, int out_size, void* d_ws,
                              size_t ws_size, hipStream_t stream) {
  const float* query = (const float*)d_in[0];
  const float* key   = (const float*)d_in[1];
  const float* value = (const float*)d_in[2];
  const float* in_w  = (const float*)d_in[3];
  const float* in_b  = (const float*)d_in[4];
  const float* out_w = (const float*)d_in[5];
  const float* out_b = (const float*)d_in[6];
  float* out = (float*)d_out;

  const size_t n = (size_t)BB * HH * SS * DD;   // 4M elems
  __bf16* qws = (__bf16*)d_ws;         // [B,H,S,D]
  __bf16* kws = qws + n;               // [B,H,S,D]
  __bf16* vtw = kws + n;               // [B,H,D,S]
  __bf16* ow  = vtw + n;               // [BS,EE]
  __bf16* xq  = ow  + n;               // bf16 copies of inputs / weights
  __bf16* xk  = xq  + n;
  __bf16* xv  = xk  + n;
  __bf16* wib = xv  + n;               // 3*EE*EE
  __bf16* wob = wib + (size_t)3 * EE * EE;

  dim3 blk(256);
  const int nx = (int)(n / 8);
  k_cvt<<<dim3(nx / 256), blk, 0, stream>>>(query, xq, nx);
  k_cvt<<<dim3(nx / 256), blk, 0, stream>>>(key,   xk, nx);
  k_cvt<<<dim3(nx / 256), blk, 0, stream>>>(value, xv, nx);
  const int nwi = 3 * EE * EE / 8;
  k_cvt<<<dim3(nwi / 256), blk, 0, stream>>>(in_w,  wib, nwi);
  const int nwo = EE * EE / 8;
  k_cvt<<<dim3(nwo / 256), blk, 0, stream>>>(out_w, wob, nwo);

  k_proj_qk<<<dim3(BS / 256, EE / 64, 2), blk, 0, stream>>>(
      xq, xk, wib, in_b, qws, kws);
  k_proj_v<<<dim3(EE / 256, BS / 64), blk, 0, stream>>>(xv, wib, in_b, vtw);
  k_attn<<<dim3(SS / 256, HH, BB), blk, 0, stream>>>(qws, kws, vtw, ow);
  k_outproj<<<dim3(BS / 256, EE / 64), blk, 0, stream>>>(ow, wob, out_b, out);
}
